// CriticEncoder_6459630813249
// MI455X (gfx1250) — compile-verified, matrix operand order checked by reading, not run on hardware
//
#include <hip/hip_runtime.h>
#include <hip/hip_fp16.h>

// ---------------- model dims ----------------
#define BATCH  32
#define EDIM   256
#define HDIM   512
#define SEQ    512
#define G4     2048      // 4*HDIM (i,f,g,o gate pre-activations)
#define NHEAD  8
#define HDHEAD 64
#define NBLK_SCAN 32     // persistent workgroups in the recurrent scan

typedef __attribute__((ext_vector_type(16))) _Float16 v16h;
typedef __attribute__((ext_vector_type(8)))  float    v8f;

union Frag16 { v16h h; uint4 u[2]; };

// Load a 16x32 f16 WMMA fragment (A layout; B uses the mirrored column layout)
// from a row-major matrix. Per the CDNA5 ISA 16-bit A layout:
//   lanes 0-15  : row = lane,     K = {0..7, 16..23}
//   lanes 16-31 : row = lane-16,  K = {8..15, 24..31}
// -> two contiguous 16-byte chunks per lane (global_load_b128 x2).
__device__ __forceinline__
v16h load_frag16(const _Float16* __restrict__ base, int row_stride, int lane) {
  const int r  = lane & 15;
  const int ko = (lane < 16) ? 0 : 8;
  const _Float16* p = base + (size_t)r * row_stride + ko;
  Frag16 f;
  f.u[0] = *(const uint4*)(p);        // K = ko .. ko+7
  f.u[1] = *(const uint4*)(p + 16);   // K = ko+16 .. ko+23
  return f.h;
}

__device__ __forceinline__
v8f wmma16(v16h a, v16h b, v8f c) {
  // D(16x16,f32) = A(16x32,f16) * B(32x16,f16) + C
  return __builtin_amdgcn_wmma_f32_16x16x32_f16(false, a, false, b, (short)0, c,
                                                false, false);
}

__device__ __forceinline__ float sigf(float x) { return 1.f / (1.f + __expf(-x)); }

// ---------------- Tensor Data Mover (TDM) prefetch ----------------
#if defined(__has_builtin)
#if __has_builtin(__builtin_amdgcn_tensor_load_to_lds)
#define ATHENA_HAVE_TDM 1
#endif
#endif

#ifdef ATHENA_HAVE_TDM
typedef unsigned int t_u32x4 __attribute__((ext_vector_type(4)));
typedef int          t_i32x4 __attribute__((ext_vector_type(4)));
typedef int          t_i32x8 __attribute__((ext_vector_type(8)));

// Issue a 3D TDM tile load:  x = 16 f32 (one gate chunk), y = 4 gates
// (stride 512 f32), z = 32 batch rows (stride 2048 f32) -> 8 KB LDS tile.
// D# field packing per CDNA5 ISA.
__device__ __forceinline__
void tdm_load_preact_tile(const float* gsrc, unsigned lds_addr) {
  unsigned long long ga = (unsigned long long)(size_t)gsrc;
  t_u32x4 g0;
  g0[0] = 1u;                                          // count=1, user mode
  g0[1] = lds_addr;                                    // LDS byte address
  g0[2] = (unsigned)(ga & 0xffffffffu);                // global_addr[31:0]
  g0[3] = (unsigned)((ga >> 32) & 0x01ffffffu)         // global_addr[56:32]
        | (2u << 30);                                  // type = 2 ("image")
  t_i32x8 g1;
  g1[0] = (int)(2u << 16);                             // data_size=4B, wg_mask=0
  g1[1] = (int)(16u << 16);                            // tensor_dim0 = 16
  g1[2] = (int)(4u << 16);                             // tensor_dim1 = 4
  g1[3] = (int)(16u << 16);                            // tile_dim0 = 16
  g1[4] = (int)(4u | (32u << 16));                     // tile_dim1=4, tile_dim2=32
  g1[5] = 512;                                         // dim0_stride = 512 f32
  g1[6] = (int)(2048u << 16);                          // dim1_stride = 2048 f32
  g1[7] = 0;
  t_i32x4 g2;
  g2[0] = 32; g2[1] = 0; g2[2] = 0; g2[3] = 0;         // tensor_dim2 = 32
  t_i32x4 g3;
  g3[0] = 0; g3[1] = 0; g3[2] = 0; g3[3] = 0;
#if __has_include(<hip/amd_detail/amd_gfx1250_TDM.h>)
  t_i32x8 g4;
  g4[0] = 0; g4[1] = 0; g4[2] = 0; g4[3] = 0;
  g4[4] = 0; g4[5] = 0; g4[6] = 0; g4[7] = 0;
  __builtin_amdgcn_tensor_load_to_lds(g0, g1, g2, g3, g4, 0);   // clang-23 form
#else
  __builtin_amdgcn_tensor_load_to_lds(g0, g1, g2, g3, 0);       // ROCm 7.2 form
#endif
}
#endif // ATHENA_HAVE_TDM

// ---------------- elementwise prep kernels ----------------
__global__ void k_cast_h(const float* __restrict__ s, _Float16* __restrict__ d, int n) {
  int i = blockIdx.x * 256 + threadIdx.x;
  if (i < n) d[i] = (_Float16)s[i];
}

// x [B,L,E] f32 -> Xall [L,B,E] f16  (rows become (t,b) pairs for the GEMM)
__global__ void k_cast_x(const float* __restrict__ x, _Float16* __restrict__ X) {
  int i = blockIdx.x * 256 + threadIdx.x;
  if (i >= BATCH * SEQ * EDIM) return;
  int e = i & (EDIM - 1);
  int l = (i >> 8) & (SEQ - 1);
  int b = i >> 17;
  X[((size_t)l * BATCH + b) * EDIM + e] = (_Float16)x[i];
}

__global__ void k_bias2(const float* __restrict__ a, const float* __restrict__ b,
                        float* __restrict__ d, int n) {
  int i = blockIdx.x * 256 + threadIdx.x;
  if (i < n) d[i] = a[i] + b[i];
}

__global__ void k_zero(unsigned long long* __restrict__ p, int n64) {
  int i = blockIdx.x * 256 + threadIdx.x;
  if (i < n64) p[i] = 0ull;
}

// ---------------- macro-tiled WMMA GEMM ----------------
// C[M,N] = A[M,K] * Bw[N,K]^T (+bias). A, Bw row-major f16.
// Block = 8 waves covering a 64x256 tile; each wave owns a 32x64 macro-tile
// (2 A-frags x 4 B-frags -> 8 WMMAs per 32-deep K-chunk; 1.5 b128 loads/WMMA).
__global__ void k_gemm(const _Float16* __restrict__ A, const _Float16* __restrict__ Bw,
                       const float* __restrict__ bias, float* __restrict__ Cf,
                       _Float16* __restrict__ Ch, int M, int N, int K) {
  const int lane = threadIdx.x & 31;
  const int wave = threadIdx.x >> 5;
  const int wm = wave & 1;                      // 2 waves along M
  const int wn = wave >> 1;                     // 4 waves along N
  const int m0 = blockIdx.x * 64 + wm * 32;     // wave rows  m0 .. m0+31
  const int n0 = blockIdx.y * 256 + wn * 64;    // wave cols  n0 .. n0+63
  if (m0 >= M || n0 >= N) return;
  v8f acc[8] = {};
  const _Float16* A0 = A + (size_t)m0 * K;
  const _Float16* A1 = A + (size_t)(m0 + 16) * K;
  for (int k0 = 0; k0 < K; k0 += 32) {
    if (k0 + 32 < K) {   // global_prefetch_b8 of next K-chunk
      __builtin_prefetch(A0 + (size_t)(lane & 15) * K + k0 + 32, 0, 1);
      __builtin_prefetch(Bw + (size_t)(n0 + (lane & 15)) * K + k0 + 32, 0, 1);
    }
    v16h a0 = load_frag16(A0 + k0, K, lane);
    v16h a1 = load_frag16(A1 + k0, K, lane);
    v16h b[4];
#pragma unroll
    for (int j = 0; j < 4; ++j)
      b[j] = load_frag16(Bw + (size_t)(n0 + 16 * j) * K + k0, K, lane);
#pragma unroll
    for (int j = 0; j < 4; ++j) {
      acc[j]     = wmma16(a0, b[j], acc[j]);
      acc[4 + j] = wmma16(a1, b[j], acc[4 + j]);
    }
  }
  const int col = lane & 15;
  const int rb  = (lane < 16) ? 0 : 8;          // C layout: vgpr r = rows {r, r+8}
#pragma unroll
  for (int i = 0; i < 2; ++i)
#pragma unroll
    for (int j = 0; j < 4; ++j) {
      const int n = n0 + 16 * j + col;
      const float bc = bias ? bias[n] : 0.f;
#pragma unroll
      for (int r = 0; r < 8; ++r) {
        float v = acc[i * 4 + j][r] + bc;
        size_t idx = (size_t)(m0 + 16 * i + rb + r) * N + n;
        if (Cf) Cf[idx] = v;
        if (Ch) Ch[idx] = (_Float16)v;
      }
    }
}

// ---------------- persistent recurrent LSTM scan ----------------
// 32 blocks; block owns a 16-unit hidden slice (its i/f/g/o columns).
// Weight-stationary: each wave keeps its 16x512 Whh slice as 16 resident
// B-fragments (128 VGPRs) across all 512 steps -> zero per-step weight traffic.
// The per-step preact gate tile is prefetched one step ahead by the Tensor
// Data Mover into a double-buffered LDS tile (TENSORcnt-tracked), hiding its
// latency behind the WMMAs + grid barrier of the current step.
__global__ void k_lstm_scan(const float* __restrict__ preact,  // [L,B,2048] f32
                            const _Float16* __restrict__ Whh,  // [2048,512] f16
                            const float* __restrict__ bias,    // bih+bhh [2048]
                            _Float16* __restrict__ hbuf,       // [2][B*H] f16
                            float* __restrict__ cbuf,          // [B*H] f32
                            _Float16* __restrict__ HoutH,      // [L,B,H] f16
                            float* __restrict__ HoutF,         // [B,L,H] f32 or null
                            unsigned int* __restrict__ bar, int nblocks) {
  __shared__ float pre_lds[2][BATCH][64];   // 2 x 8 KB TDM-staged gate tiles
  __shared__ float gl[2][4][16][16];        // gate tiles: [mtile][gate][row][col]
  const int lane = threadIdx.x & 31;
  const int wave = threadIdx.x >> 5;
  const int mt   = wave & 1;                // batch tile (rows 0-15 / 16-31)
  const int gate = wave >> 1;               // i,f,g,o
  const int hs   = blockIdx.x * 16;         // this block's hidden slice
  const int gcol0 = gate * HDIM + hs;
  const int col = lane & 15;
  const int rb  = (lane < 16) ? 0 : 8;
  // hoist the weight slice into registers (L2 -> VGPR, once)
  v16h bw[16];
  {
    const _Float16* Bb = Whh + (size_t)gcol0 * HDIM;
#pragma unroll
    for (int k = 0; k < 16; ++k) bw[k] = load_frag16(Bb + k * 32, HDIM, lane);
  }
#ifdef ATHENA_HAVE_TDM
  // prime: TDM load of the t=0 tile into buffer 0 (wave 0 issues; EXEC ignored)
  if (wave == 0)
    tdm_load_preact_tile(preact + (size_t)0 * BATCH * G4 + hs,
                         (unsigned)(size_t)&pre_lds[0][0][0]);
#endif
  unsigned int target = 0;
  for (int t = 0; t < SEQ; ++t) {
    const _Float16* hread  = hbuf + (t & 1) * (BATCH * HDIM);
    _Float16*       hwrite = hbuf + ((t + 1) & 1) * (BATCH * HDIM);
#ifdef ATHENA_HAVE_TDM
    if (wave == 0) {
      if (t + 1 < SEQ) {
        // issue next tile, then wait for the *current* one (in-order TENSORcnt)
        tdm_load_preact_tile(preact + (size_t)(t + 1) * BATCH * G4 + hs,
                             (unsigned)(size_t)&pre_lds[(t + 1) & 1][0][0]);
        __builtin_amdgcn_s_wait_tensorcnt(1);
      } else {
        __builtin_amdgcn_s_wait_tensorcnt(0);
      }
    }
    __syncthreads();                        // publish staged tile block-wide
#else
    // fallback: cooperative global->LDS copy of the step-t gate tile
    {
      const float* P = preact + (size_t)t * BATCH * G4 + hs;
      for (int e = threadIdx.x; e < BATCH * 64; e += blockDim.x) {
        int z = e >> 6, y = (e >> 4) & 3, xx = e & 15;
        pre_lds[t & 1][z][(y << 4) | xx] = P[(size_t)z * G4 + y * HDIM + xx];
      }
    }
    __syncthreads();
#endif
    // init accumulator with the staged input projection + bias (LDS reads)
    v8f acc;
    {
      const float bc = bias[gcol0 + col];
#pragma unroll
      for (int r = 0; r < 8; ++r)
        acc[r] = pre_lds[t & 1][mt * 16 + rb + r][(gate << 4) | col] + bc;
    }
    // gates += h_{t-1} @ Whh^T : 16 WMMAs, A-frags 2-deep software pipelined
    const _Float16* Ab = hread + (size_t)(mt * 16) * HDIM;
    v16h a_cur = load_frag16(Ab, HDIM, lane);
#pragma unroll
    for (int k = 0; k < 16; ++k) {
      v16h a_nxt = a_cur;
      if (k < 15) a_nxt = load_frag16(Ab + (k + 1) * 32, HDIM, lane);
      acc = wmma16(a_cur, bw[k], acc);
      a_cur = a_nxt;
    }
#pragma unroll
    for (int r = 0; r < 8; ++r) gl[mt][gate][rb + r][col] = acc[r];
    __syncthreads();
    // cell update for this block's slice (32 batch x 16 units)
    for (int e = threadIdx.x; e < BATCH * 16; e += blockDim.x) {
      int bI = e >> 4;
      int c  = e & 15;
      float gi = sigf (gl[bI >> 4][0][bI & 15][c]);
      float gf = sigf (gl[bI >> 4][1][bI & 15][c]);
      float gg = tanhf(gl[bI >> 4][2][bI & 15][c]);
      float go = sigf (gl[bI >> 4][3][bI & 15][c]);
      float cp = cbuf[bI * HDIM + hs + c];
      float cn = gf * cp + gi * gg;
      float hn = go * tanhf(cn);
      cbuf[bI * HDIM + hs + c]   = cn;
      hwrite[bI * HDIM + hs + c] = (_Float16)hn;
      HoutH[((size_t)t * BATCH + bI) * HDIM + hs + c] = (_Float16)hn;
      if (HoutF) HoutF[((size_t)bI * SEQ + t) * HDIM + hs + c] = hn;
    }
    // device-scope grid barrier (monotonic counter, acquire/release)
    __threadfence();
    __syncthreads();
    target += (unsigned)nblocks;
    if (threadIdx.x == 0) {
      __hip_atomic_fetch_add(bar, 1u, __ATOMIC_RELEASE, __HIP_MEMORY_SCOPE_AGENT);
      while (__hip_atomic_load(bar, __ATOMIC_ACQUIRE, __HIP_MEMORY_SCOPE_AGENT) < target)
        __builtin_amdgcn_s_sleep(1);
    }
    __syncthreads();
  }
}

// ---------------- attention weights (scores+softmax+head-mean) ----------------
// block = (batch b, 16-query tile). Scores per head via WMMA (K=64 -> 2 WMMAs),
// q fragments hoisted out of the key loop; causal mask; softmax over LDS row
// buffer; mean over heads accumulated in registers.
__global__ void k_attn(const _Float16* __restrict__ q,   // [L,B,H] f16
                       const _Float16* __restrict__ kk,  // [L,B,H] f16
                       float* __restrict__ attn) {       // [B,L,L] f32
  __shared__ float sc[16][SEQ];          // 32 KB score rows
  const int b    = blockIdx.x;
  const int qt   = blockIdx.y;
  const int lane = threadIdx.x & 31;
  const int wave = threadIdx.x >> 5;
  const int row  = threadIdx.x >> 4;     // softmax: 16 groups of 16 threads
  const int sub  = threadIdx.x & 15;
  const int col  = lane & 15;
  const int rb   = (lane < 16) ? 0 : 8;
  float accm[32];
#pragma unroll
  for (int j = 0; j < 32; ++j) accm[j] = 0.f;
  for (int h = 0; h < NHEAD; ++h) {
    const _Float16* qb = q + ((size_t)(qt * 16) * BATCH + b) * HDIM + h * HDHEAD;
    v16h aq0 = load_frag16(qb, BATCH * HDIM, lane);        // K = 0..31 of head
    v16h aq1 = load_frag16(qb + 32, BATCH * HDIM, lane);   // K = 32..63
    for (int kt = wave; kt < SEQ / 16; kt += 8) {
      const _Float16* kb = kk + ((size_t)(kt * 16) * BATCH + b) * HDIM + h * HDHEAD;
      v16h b0 = load_frag16(kb, BATCH * HDIM, lane);
      v16h b1 = load_frag16(kb + 32, BATCH * HDIM, lane);
      v8f acc = {};
      acc = wmma16(aq0, b0, acc);
      acc = wmma16(aq1, b1, acc);
#pragma unroll
      for (int r = 0; r < 8; ++r) {
        int qrow = qt * 16 + rb + r;
        int n    = kt * 16 + col;
        float v  = acc[r] * 0.125f;            // 1/sqrt(64)
        if (n > qrow) v = -1e30f;              // causal mask
        sc[rb + r][n] = v;
      }
    }
    __syncthreads();
    float mx = -1e30f;
    for (int j = 0; j < 32; ++j) mx = fmaxf(mx, sc[row][sub + 16 * j]);
    for (int off = 8; off >= 1; off >>= 1) mx = fmaxf(mx, __shfl_xor(mx, off, 16));
    float sm = 0.f;
    for (int j = 0; j < 32; ++j) sm += __expf(sc[row][sub + 16 * j] - mx);
    for (int off = 8; off >= 1; off >>= 1) sm += __shfl_xor(sm, off, 16);
    float inv = 0.125f / sm;                   // softmax * (1/NHEAD)
    for (int j = 0; j < 32; ++j) accm[j] += __expf(sc[row][sub + 16 * j] - mx) * inv;
    __syncthreads();
  }
  for (int j = 0; j < 32; ++j)
    attn[((size_t)b * SEQ + qt * 16 + row) * SEQ + sub + 16 * j] = accm[j];
}

// ---------------- fused gate + final linear ----------------
// out[b,l] = bo + sum_n attn[b,l,n] * h1[b,l,n] * Wo[n]   (one wave per row)
__global__ void k_out(const float* __restrict__ attn, const float* __restrict__ h1f,
                      const float* __restrict__ Wo, const float* __restrict__ bo,
                      float* __restrict__ out) {
  int idx  = blockIdx.x * 8 + (threadIdx.x >> 5);     // 0..16383
  int lane = threadIdx.x & 31;
  const float* ar = attn + (size_t)idx * SEQ;
  const float* hr = h1f  + (size_t)idx * HDIM;
  float s = 0.f;
  for (int n = lane; n < HDIM; n += 32) s += ar[n] * hr[n] * Wo[n];
  for (int off = 16; off >= 1; off >>= 1) s += __shfl_down(s, off, 32);
  if (lane == 0) out[idx] = s + bo[0];
}

// ---------------- host orchestration ----------------
static inline int cdiv_h(int a, int b) { return (a + b - 1) / b; }

extern "C" void kernel_launch(void* const* d_in, const int* in_sizes, int n_in,
                              void* d_out, int out_size, void* d_ws, size_t ws_size,
                              hipStream_t stream) {
  (void)in_sizes; (void)n_in; (void)out_size; (void)ws_size;
  const float* x    = (const float*)d_in[0];
  const float* Wih0 = (const float*)d_in[1];
  const float* Whh0 = (const float*)d_in[2];
  const float* bih0 = (const float*)d_in[3];
  const float* bhh0 = (const float*)d_in[4];
  const float* Wih1 = (const float*)d_in[5];
  const float* Whh1 = (const float*)d_in[6];
  const float* bih1 = (const float*)d_in[7];
  const float* bhh1 = (const float*)d_in[8];
  const float* Wq   = (const float*)d_in[9];
  const float* bq   = (const float*)d_in[10];
  const float* Wk   = (const float*)d_in[11];
  const float* bk   = (const float*)d_in[12];
  const float* Wo   = (const float*)d_in[13];
  const float* bo   = (const float*)d_in[14];
  float* out = (float*)d_out;

  char* wsp = (char*)d_ws;
  size_t off = 0;
  auto take = [&](size_t bytes) -> void* {
    void* p = wsp + off;
    off += (bytes + 255) & ~(size_t)255;
    return p;
  };
  float*     preact = (float*)    take((size_t)SEQ * BATCH * G4 * 4);     // 128 MB (reused by both layers)
  _Float16*  Xall   = (_Float16*) take((size_t)SEQ * BATCH * EDIM * 2);
  _Float16*  H0h    = (_Float16*) take((size_t)SEQ * BATCH * HDIM * 2);
  _Float16*  H1h    = (_Float16*) take((size_t)SEQ * BATCH * HDIM * 2);
  float*     H1f    = (float*)    take((size_t)BATCH * SEQ * HDIM * 4);
  _Float16*  qh     = (_Float16*) take((size_t)SEQ * BATCH * HDIM * 2);
  _Float16*  kh     = (_Float16*) take((size_t)SEQ * BATCH * HDIM * 2);
  float*     attn   = (float*)    take((size_t)BATCH * SEQ * SEQ * 4);
  _Float16*  Wih0h  = (_Float16*) take((size_t)G4 * EDIM * 2);
  _Float16*  Whh0h  = (_Float16*) take((size_t)G4 * HDIM * 2);
  _Float16*  Wih1h  = (_Float16*) take((size_t)G4 * HDIM * 2);
  _Float16*  Whh1h  = (_Float16*) take((size_t)G4 * HDIM * 2);
  _Float16*  Wqh    = (_Float16*) take((size_t)HDIM * HDIM * 2);
  _Float16*  Wkh    = (_Float16*) take((size_t)HDIM * HDIM * 2);
  float*     bias0  = (float*)    take(G4 * 4);
  float*     bias1  = (float*)    take(G4 * 4);
  // contiguous scan-state block (zeroed together each launch)
  _Float16*  hstate = (_Float16*) take(2 * BATCH * HDIM * 2);   // 64 KB
  float*     cstate = (float*)    take(BATCH * HDIM * 4);       // 64 KB
  unsigned int* bar = (unsigned int*)take(256);
  const int state_n64 = (2 * BATCH * HDIM * 2 + BATCH * HDIM * 4 + 256) / 8;

  // weight/input casts to f16
  k_cast_x<<<cdiv_h(BATCH * SEQ * EDIM, 256), 256, 0, stream>>>(x, Xall);
  k_cast_h<<<cdiv_h(G4 * EDIM, 256), 256, 0, stream>>>(Wih0, Wih0h, G4 * EDIM);
  k_cast_h<<<cdiv_h(G4 * HDIM, 256), 256, 0, stream>>>(Whh0, Whh0h, G4 * HDIM);
  k_cast_h<<<cdiv_h(G4 * HDIM, 256), 256, 0, stream>>>(Wih1, Wih1h, G4 * HDIM);
  k_cast_h<<<cdiv_h(G4 * HDIM, 256), 256, 0, stream>>>(Whh1, Whh1h, G4 * HDIM);
  k_cast_h<<<cdiv_h(HDIM * HDIM, 256), 256, 0, stream>>>(Wq, Wqh, HDIM * HDIM);
  k_cast_h<<<cdiv_h(HDIM * HDIM, 256), 256, 0, stream>>>(Wk, Wkh, HDIM * HDIM);
  k_bias2<<<8, 256, 0, stream>>>(bih0, bhh0, bias0, G4);
  k_bias2<<<8, 256, 0, stream>>>(bih1, bhh1, bias1, G4);

  // layer 0: hoisted input projection, then persistent scan
  k_gemm<<<dim3(SEQ * BATCH / 64, G4 / 256), 256, 0, stream>>>(
      Xall, Wih0h, nullptr, preact, nullptr, SEQ * BATCH, G4, EDIM);
  k_zero<<<cdiv_h(state_n64, 256), 256, 0, stream>>>((unsigned long long*)hstate, state_n64);
  k_lstm_scan<<<NBLK_SCAN, 256, 0, stream>>>(preact, Whh0h, bias0, hstate, cstate,
                                             H0h, nullptr, bar, NBLK_SCAN);

  // layer 1
  k_gemm<<<dim3(SEQ * BATCH / 64, G4 / 256), 256, 0, stream>>>(
      H0h, Wih1h, nullptr, preact, nullptr, SEQ * BATCH, G4, HDIM);
  k_zero<<<cdiv_h(state_n64, 256), 256, 0, stream>>>((unsigned long long*)hstate, state_n64);
  k_lstm_scan<<<NBLK_SCAN, 256, 0, stream>>>(preact, Whh1h, bias1, hstate, cstate,
                                             H1h, H1f, bar, NBLK_SCAN);

  // q/k projections (f16 outputs for the score WMMAs)
  k_gemm<<<dim3(SEQ * BATCH / 64, HDIM / 256), 256, 0, stream>>>(
      H1h, Wqh, bq, nullptr, qh, SEQ * BATCH, HDIM, HDIM);
  k_gemm<<<dim3(SEQ * BATCH / 64, HDIM / 256), 256, 0, stream>>>(
      H1h, Wkh, bk, nullptr, kh, SEQ * BATCH, HDIM, HDIM);

  // attention weights, then fused gate + output linear
  k_attn<<<dim3(BATCH, SEQ / 16), 256, 0, stream>>>(qh, kh, attn);
  k_out<<<BATCH * SEQ / 8, 256, 0, stream>>>(attn, H1f, Wo, bo, out);
}
